// MessagePassing_12979391168960
// MI455X (gfx1250) — compile-verified, matrix operand order checked by reading, not run on hardware
//
#include <hip/hip_runtime.h>
#include <cstdint>

// ---------------------------------------------------------------------------
// MessagePassing on MI455X (gfx1250, wave32, WMMA bf16 split-precision)
// ---------------------------------------------------------------------------

typedef __attribute__((ext_vector_type(16))) __bf16 v16bf;
typedef __attribute__((ext_vector_type(8)))  __bf16 v8bf;
typedef __attribute__((ext_vector_type(8)))  float  v8f;

#define NINST 4096
#define NPHRA 2048
#define NEDGE 65536
#define DIM   1024
#define DOUT  128

// ---------------------------------------------------------------------------
// fp32 -> (bf16 hi, bf16 lo) split conversion
// ---------------------------------------------------------------------------
__global__ void cvt_split_kernel(const float* __restrict__ x,
                                 __bf16* __restrict__ hi,
                                 __bf16* __restrict__ lo, int n) {
  int i = blockIdx.x * blockDim.x + threadIdx.x;
  if (i < n) {
    float v = x[i];
    __bf16 h = (__bf16)v;
    hi[i] = h;
    lo[i] = (__bf16)(v - (float)h);
  }
}

// Weight transpose + split: W[k][n] (row-major KxNc) -> Wt[n][k] bf16 hi/lo
__global__ void wsplit_t_kernel(const float* __restrict__ w,
                                __bf16* __restrict__ hi,
                                __bf16* __restrict__ lo,
                                int K, int Nc) {
  int i = blockIdx.x * blockDim.x + threadIdx.x;
  if (i < K * Nc) {
    int k = i / Nc, n = i % Nc;
    float v = w[i];
    __bf16 h = (__bf16)v;
    size_t o = (size_t)n * K + k;
    hi[o] = h;
    lo[o] = (__bf16)(v - (float)h);
  }
}

__global__ void zero_f32_kernel(float* __restrict__ p, long n) {
  long i = (long)blockIdx.x * blockDim.x + threadIdx.x;
  long stride = (long)gridDim.x * blockDim.x;
  for (; i < n; i += stride) p[i] = 0.0f;
}

__global__ void count_kernel(const int* __restrict__ idx, int* __restrict__ cnt, int n) {
  int i = blockIdx.x * blockDim.x + threadIdx.x;
  if (i < n)
    __hip_atomic_fetch_add(&cnt[idx[i]], 1, __ATOMIC_RELAXED, __HIP_MEMORY_SCOPE_AGENT);
}

// ---------------------------------------------------------------------------
// WMMA fragment load: 16-bit A/B 16x32 layout (ISA 7.12.2)
//  lane half=0: elements i<8 -> K=k0+i,    i>=8 -> K=k0+16+(i-8)
//  lane half=1: same +8
// -> two contiguous 16B chunks per lane.
// ---------------------------------------------------------------------------
__device__ __forceinline__ v16bf load_frag(const __bf16* __restrict__ row, int k0, int half) {
  const __bf16* p = row + k0 + 8 * half;
  v8bf c0 = *(const v8bf*)(p);
  v8bf c1 = *(const v8bf*)(p + 16);
  v16bf f;
#pragma unroll
  for (int i = 0; i < 8; ++i) { f[i] = c0[i]; f[i + 8] = c1[i]; }
  return f;
}

// ---------------------------------------------------------------------------
// Split-precision bf16 WMMA GEMM: C[MxNc] (op)= A[MxK] * B^T[NcxK]
//   A: bf16 hi/lo row-major, row stride = K
//   B: bf16 hi/lo row-major [Nc][ldb], k offset kb
//   mode 0: C = acc           (projection GEMMs)
//   mode 1: C += relu(acc+b)  (refine GEMMs, accumulate onto msg in d_out)
// Each wave: 16x64 strip of C (4 tiles). Block = 4 waves = 64x64.
// ---------------------------------------------------------------------------
__global__ void wmma_gemm_kernel(const __bf16* __restrict__ Ahi, const __bf16* __restrict__ Alo,
                                 const __bf16* __restrict__ Bhi, const __bf16* __restrict__ Blo,
                                 const float* __restrict__ bias,
                                 float* __restrict__ C,
                                 int K, int ldb, int kb, int ldc, int mode) {
  int lane = threadIdx.x & 31;
  int wave = threadIdx.x >> 5;
  int half = lane >> 4;
  int r    = lane & 15;
  int m     = (blockIdx.y * 4 + wave) * 16;
  int nbase = blockIdx.x * 64;

  const __bf16* arow_h = Ahi + (size_t)(m + r) * K;
  const __bf16* arow_l = Alo + (size_t)(m + r) * K;
  const __bf16* brow_h[4];
  const __bf16* brow_l[4];
#pragma unroll
  for (int t = 0; t < 4; ++t) {
    size_t off = (size_t)(nbase + t * 16 + r) * ldb + kb;
    brow_h[t] = Bhi + off;
    brow_l[t] = Blo + off;
  }

  v8f acc[4] = {};
  for (int k0 = 0; k0 < K; k0 += 32) {
    v16bf ah = load_frag(arow_h, k0, half);
    v16bf al = load_frag(arow_l, k0, half);
#pragma unroll
    for (int t = 0; t < 4; ++t) {
      v16bf bh = load_frag(brow_h[t], k0, half);
      v16bf bl = load_frag(brow_l[t], k0, half);
      // a*b ~= ah*bh + ah*bl + al*bh  (error-compensated bf16 split)
      acc[t] = __builtin_amdgcn_wmma_f32_16x16x32_bf16(false, ah, false, bh, (short)0, acc[t], false, false);
      acc[t] = __builtin_amdgcn_wmma_f32_16x16x32_bf16(false, ah, false, bl, (short)0, acc[t], false, false);
      acc[t] = __builtin_amdgcn_wmma_f32_16x16x32_bf16(false, al, false, bh, (short)0, acc[t], false, false);
    }
  }

  // C/D layout: vgpr v -> row m + v + 8*half, col nbase + t*16 + r
#pragma unroll
  for (int t = 0; t < 4; ++t) {
    int col = nbase + t * 16 + r;
    float b = (mode == 1) ? bias[col] : 0.0f;
#pragma unroll
    for (int v = 0; v < 8; ++v) {
      size_t idx = (size_t)(m + v + 8 * half) * ldc + col;
      if (mode == 1) C[idx] += fmaxf(acc[t][v] + b, 0.0f);
      else           C[idx] = acc[t][v];
    }
  }
}

// ---------------------------------------------------------------------------
// Gate kernel: one wave per edge. gate = mean_j sigmoid(relu(A[.]+B[.]+b))
// Folds the (p_o+p_s)*0.5 factor and segment-mean 1/count into the stored weight.
// ---------------------------------------------------------------------------
__global__ void gate_kernel(const int* __restrict__ sidx, const int* __restrict__ oidx,
                            const int* __restrict__ cidx,
                            const float* __restrict__ pA1, const float* __restrict__ pB1,
                            const float* __restrict__ pA2, const float* __restrict__ pB2,
                            const float* __restrict__ pA3, const float* __restrict__ pB3,
                            const float* __restrict__ pA4, const float* __restrict__ pB4,
                            const float* __restrict__ b1, const float* __restrict__ b2,
                            const float* __restrict__ b3, const float* __restrict__ b4,
                            const int* __restrict__ cnt_s, const int* __restrict__ cnt_o,
                            const int* __restrict__ cnt_c,
                            float* __restrict__ gw) {
  int e = (int)((blockIdx.x * blockDim.x + threadIdx.x) >> 5);
  int lane = threadIdx.x & 31;
  if (e >= NEDGE) return;
  int s = sidx[e], o = oidx[e], c = cidx[e];
  const float* a1 = pA1 + (size_t)c * DOUT; const float* t1 = pB1 + (size_t)s * DOUT;
  const float* a2 = pA2 + (size_t)c * DOUT; const float* t2 = pB2 + (size_t)o * DOUT;
  const float* a3 = pA3 + (size_t)o * DOUT; const float* t3 = pB3 + (size_t)c * DOUT;
  const float* a4 = pA4 + (size_t)s * DOUT; const float* t4 = pB4 + (size_t)c * DOUT;
  float s1 = 0.f, s2 = 0.f, s3 = 0.f, s4 = 0.f;
  for (int j = lane; j < DOUT; j += 32) {
    float x1 = fmaxf(a1[j] + t1[j] + b1[j], 0.f); s1 += 1.f / (1.f + __expf(-x1));
    float x2 = fmaxf(a2[j] + t2[j] + b2[j], 0.f); s2 += 1.f / (1.f + __expf(-x2));
    float x3 = fmaxf(a3[j] + t3[j] + b3[j], 0.f); s3 += 1.f / (1.f + __expf(-x3));
    float x4 = fmaxf(a4[j] + t4[j] + b4[j], 0.f); s4 += 1.f / (1.f + __expf(-x4));
  }
#pragma unroll
  for (int off = 16; off > 0; off >>= 1) {
    s1 += __shfl_xor(s1, off, 32);
    s2 += __shfl_xor(s2, off, 32);
    s3 += __shfl_xor(s3, off, 32);
    s4 += __shfl_xor(s4, off, 32);
  }
  if (lane == 0) {
    const float inv = 0.5f / (float)DOUT;  // 0.5 * mean over OUT
    gw[4 * e + 0] = s1 * inv / fmaxf((float)cnt_s[s], 1.f);
    gw[4 * e + 1] = s2 * inv / fmaxf((float)cnt_o[o], 1.f);
    float ic = inv / fmaxf((float)cnt_c[c], 1.f);
    gw[4 * e + 2] = s3 * ic;
    gw[4 * e + 3] = s4 * ic;
  }
}

// ---------------------------------------------------------------------------
// Scatter: one block per edge, fp32 atomic accumulation of scaled messages.
// ---------------------------------------------------------------------------
__global__ void scatter_kernel(const float* __restrict__ inst, const float* __restrict__ phra,
                               const int* __restrict__ sidx, const int* __restrict__ oidx,
                               const int* __restrict__ cidx,
                               const float* __restrict__ gw,
                               float* __restrict__ accI, float* __restrict__ accP) {
  int e = blockIdx.x;
  int s = sidx[e], o = oidx[e], c = cidx[e];
  float w1 = gw[4 * e + 0], w2 = gw[4 * e + 1];
  float w3 = gw[4 * e + 2], w4 = gw[4 * e + 3];
  const float* pf = phra + (size_t)c * DIM;
  const float* sf = inst + (size_t)s * DIM;
  const float* of = inst + (size_t)o * DIM;
  float* as = accI + (size_t)s * DIM;
  float* ao = accI + (size_t)o * DIM;
  float* ac = accP + (size_t)c * DIM;
  for (int d = threadIdx.x; d < DIM; d += blockDim.x) {
    float pv = pf[d];
    __hip_atomic_fetch_add(&as[d], pv * w1, __ATOMIC_RELAXED, __HIP_MEMORY_SCOPE_AGENT);
    __hip_atomic_fetch_add(&ao[d], pv * w2, __ATOMIC_RELAXED, __HIP_MEMORY_SCOPE_AGENT);
    __hip_atomic_fetch_add(&ac[d], of[d] * w3 + sf[d] * w4, __ATOMIC_RELAXED, __HIP_MEMORY_SCOPE_AGENT);
  }
}

// ---------------------------------------------------------------------------
static void launch_gemm(const __bf16* Ah, const __bf16* Al,
                        const __bf16* Bh, const __bf16* Bl,
                        const float* bias, float* C,
                        int M, int Nc, int K, int ldb, int kb, int ldc, int mode,
                        hipStream_t s) {
  dim3 grid(Nc / 64, M / 64);
  wmma_gemm_kernel<<<grid, 128, 0, s>>>(Ah, Al, Bh, Bl, bias, C, K, ldb, kb, ldc, mode);
}

extern "C" void kernel_launch(void* const* d_in, const int* in_sizes, int n_in,
                              void* d_out, int out_size, void* d_ws, size_t ws_size,
                              hipStream_t stream) {
  const float* inst = (const float*)d_in[0];
  const float* phra = (const float*)d_in[1];
  const int*   conn = (const int*)d_in[2];
  const int*   sidx = conn;
  const int*   oidx = conn + NEDGE;
  const int*   cidx = (const int*)d_in[3];
  const float* psW = (const float*)d_in[4];  const float* psB = (const float*)d_in[5];
  const float* poW = (const float*)d_in[6];  const float* poB = (const float*)d_in[7];
  const float* opW = (const float*)d_in[8];  const float* opB = (const float*)d_in[9];
  const float* spW = (const float*)d_in[10]; const float* spB = (const float*)d_in[11];
  const float* iw1W = (const float*)d_in[12]; const float* iw1B = (const float*)d_in[13];
  const float* iw2W = (const float*)d_in[14]; const float* iw2B = (const float*)d_in[15];
  const float* pw1W = (const float*)d_in[16]; const float* pw1B = (const float*)d_in[17];
  const float* pw2W = (const float*)d_in[18]; const float* pw2B = (const float*)d_in[19];

  float* outI = (float*)d_out;                  // [N, D] inst_out (also msg accumulator)
  float* outP = outI + (size_t)NINST * DIM;     // [P, D] phra_out

  // ---- workspace layout (256B aligned slices) ----
  char* cur = (char*)d_ws;
  auto alloc = [&](size_t bytes) -> char* {
    char* p = cur;
    cur += (bytes + 255) & ~(size_t)255;
    return p;
  };
  __bf16* instH = (__bf16*)alloc((size_t)NINST * DIM * 2);
  __bf16* instL = (__bf16*)alloc((size_t)NINST * DIM * 2);
  __bf16* phraH = (__bf16*)alloc((size_t)NPHRA * DIM * 2);
  __bf16* phraL = (__bf16*)alloc((size_t)NPHRA * DIM * 2);
  // gate weights transposed: [DOUT][2*DIM]
  __bf16 *gwtH[4], *gwtL[4];
  for (int i = 0; i < 4; ++i) {
    gwtH[i] = (__bf16*)alloc((size_t)2 * DIM * DOUT * 2);
    gwtL[i] = (__bf16*)alloc((size_t)2 * DIM * DOUT * 2);
  }
  // refine weights transposed: [DIM][DIM]
  __bf16 *rwtH[4], *rwtL[4];
  for (int i = 0; i < 4; ++i) {
    rwtH[i] = (__bf16*)alloc((size_t)DIM * DIM * 2);
    rwtL[i] = (__bf16*)alloc((size_t)DIM * DIM * 2);
  }
  float* pA1 = (float*)alloc((size_t)NPHRA * DOUT * 4);  // phra @ psW[0:D]
  float* pB1 = (float*)alloc((size_t)NINST * DOUT * 4);  // inst @ psW[D:2D]
  float* pA2 = (float*)alloc((size_t)NPHRA * DOUT * 4);
  float* pB2 = (float*)alloc((size_t)NINST * DOUT * 4);
  float* pA3 = (float*)alloc((size_t)NINST * DOUT * 4);  // inst @ opW[0:D]
  float* pB3 = (float*)alloc((size_t)NPHRA * DOUT * 4);
  float* pA4 = (float*)alloc((size_t)NINST * DOUT * 4);
  float* pB4 = (float*)alloc((size_t)NPHRA * DOUT * 4);
  float* gw  = (float*)alloc((size_t)NEDGE * 4 * 4);
  int*   cnt = (int*)alloc((size_t)(2 * NINST + NPHRA) * 4);
  int* cnt_s = cnt;
  int* cnt_o = cnt + NINST;
  int* cnt_c = cnt + 2 * NINST;
  __bf16* msgIH = (__bf16*)alloc((size_t)NINST * DIM * 2);
  __bf16* msgIL = (__bf16*)alloc((size_t)NINST * DIM * 2);
  __bf16* msgPH = (__bf16*)alloc((size_t)NPHRA * DIM * 2);
  __bf16* msgPL = (__bf16*)alloc((size_t)NPHRA * DIM * 2);

  // ---- phase 1: precision split of features + weights ----
  {
    int n = NINST * DIM;
    cvt_split_kernel<<<(n + 255) / 256, 256, 0, stream>>>(inst, instH, instL, n);
    n = NPHRA * DIM;
    cvt_split_kernel<<<(n + 255) / 256, 256, 0, stream>>>(phra, phraH, phraL, n);
  }
  {
    const float* gws[4] = {psW, poW, opW, spW};
    int n = 2 * DIM * DOUT;
    for (int i = 0; i < 4; ++i)
      wsplit_t_kernel<<<(n + 255) / 256, 256, 0, stream>>>(gws[i], gwtH[i], gwtL[i], 2 * DIM, DOUT);
    const float* rws[4] = {iw1W, iw2W, pw1W, pw2W};
    n = DIM * DIM;
    for (int i = 0; i < 4; ++i)
      wsplit_t_kernel<<<(n + 255) / 256, 256, 0, stream>>>(rws[i], rwtH[i], rwtL[i], DIM, DIM);
  }

  // ---- phase 2: segment counts ----
  zero_f32_kernel<<<64, 256, 0, stream>>>((float*)cnt, (long)(2 * NINST + NPHRA));
  count_kernel<<<NEDGE / 256, 256, 0, stream>>>(sidx, cnt_s, NEDGE);
  count_kernel<<<NEDGE / 256, 256, 0, stream>>>(oidx, cnt_o, NEDGE);
  count_kernel<<<NEDGE / 256, 256, 0, stream>>>(cidx, cnt_c, NEDGE);

  // ---- phase 3: node-level gate projections (8 WMMA GEMMs, mode 0) ----
  // p_s: src=phr (W rows 0:D), tgt=subj (W rows D:2D)
  launch_gemm(phraH, phraL, gwtH[0], gwtL[0], nullptr, pA1, NPHRA, DOUT, DIM, 2 * DIM, 0,    DOUT, 0, stream);
  launch_gemm(instH, instL, gwtH[0], gwtL[0], nullptr, pB1, NINST, DOUT, DIM, 2 * DIM, DIM,  DOUT, 0, stream);
  // p_o
  launch_gemm(phraH, phraL, gwtH[1], gwtL[1], nullptr, pA2, NPHRA, DOUT, DIM, 2 * DIM, 0,    DOUT, 0, stream);
  launch_gemm(instH, instL, gwtH[1], gwtL[1], nullptr, pB2, NINST, DOUT, DIM, 2 * DIM, DIM,  DOUT, 0, stream);
  // o_p: src=obj (inst), tgt=phr
  launch_gemm(instH, instL, gwtH[2], gwtL[2], nullptr, pA3, NINST, DOUT, DIM, 2 * DIM, 0,    DOUT, 0, stream);
  launch_gemm(phraH, phraL, gwtH[2], gwtL[2], nullptr, pB3, NPHRA, DOUT, DIM, 2 * DIM, DIM,  DOUT, 0, stream);
  // s_p: src=subj (inst), tgt=phr
  launch_gemm(instH, instL, gwtH[3], gwtL[3], nullptr, pA4, NINST, DOUT, DIM, 2 * DIM, 0,    DOUT, 0, stream);
  launch_gemm(phraH, phraL, gwtH[3], gwtL[3], nullptr, pB4, NPHRA, DOUT, DIM, 2 * DIM, DIM,  DOUT, 0, stream);

  // ---- phase 4: per-edge gates (wave per edge) ----
  gate_kernel<<<(NEDGE * 32) / 256, 256, 0, stream>>>(
      sidx, oidx, cidx, pA1, pB1, pA2, pB2, pA3, pB3, pA4, pB4,
      psB, poB, opB, spB, cnt_s, cnt_o, cnt_c, gw);

  // ---- phase 5: scatter messages into d_out (acts as msg accumulator) ----
  zero_f32_kernel<<<4096, 256, 0, stream>>>(outI, (long)out_size);
  scatter_kernel<<<NEDGE, 256, 0, stream>>>(inst, phra, sidx, oidx, cidx, gw, outI, outP);

  // ---- phase 6: split msg to bf16, then refine GEMMs accumulate in place ----
  {
    int n = NINST * DIM;
    cvt_split_kernel<<<(n + 255) / 256, 256, 0, stream>>>(outI, msgIH, msgIL, n);
    n = NPHRA * DIM;
    cvt_split_kernel<<<(n + 255) / 256, 256, 0, stream>>>(outP, msgPH, msgPL, n);
  }
  // out = msg + relu(msg@w1 + b1) + relu(feat@w2 + b2)
  launch_gemm(msgIH, msgIL, rwtH[0], rwtL[0], iw1B, outI, NINST, DIM, DIM, DIM, 0, DIM, 1, stream);
  launch_gemm(instH, instL, rwtH[1], rwtL[1], iw2B, outI, NINST, DIM, DIM, DIM, 0, DIM, 1, stream);
  launch_gemm(msgPH, msgPL, rwtH[2], rwtL[2], pw1B, outP, NPHRA, DIM, DIM, DIM, 0, DIM, 1, stream);
  launch_gemm(phraH, phraL, rwtH[3], rwtL[3], pw2B, outP, NPHRA, DIM, DIM, DIM, 0, DIM, 1, stream);
}
